// Nonlocal_47021301957431
// MI455X (gfx1250) — matrix operand order repaired and checked by reading, not compiled, not run on hardware
//
#include <hip/hip_runtime.h>
#include <hip/hip_bf16.h>
#include <math.h>

typedef __attribute__((ext_vector_type(16))) __bf16 v16bf;
typedef __attribute__((ext_vector_type(8)))  float  v8f;
typedef __attribute__((ext_vector_type(4)))  unsigned int u32x4;
typedef __attribute__((ext_vector_type(8)))  int i32x8;
typedef __attribute__((ext_vector_type(4)))  int i32x4;

#define B_ 8
#define C_ 256
#define I_ 128
#define N_ 4096
#define BN_EPS 1e-5f

__device__ __forceinline__ v8f wmma_bf16(v16bf a, v16bf b, v8f c) {
  return __builtin_amdgcn_wmma_f32_16x16x32_bf16(false, a, false, b, (short)0, c,
                                                 false, false);
}

__device__ __forceinline__ v8f zero_v8f() {
  v8f z;
#pragma unroll
  for (int i = 0; i < 8; ++i) z[i] = 0.0f;
  return z;
}

// A fragment (16x32 bf16, MxK). Row = lane%16; per-lane K runs per ISA 7.12.2:
// K = (j>>3)*16 + (lane>>4)*8 + (j&7). Source row-major, K contiguous.
__device__ __forceinline__ v16bf load_a_bf16(const __bf16* base, size_t ldk, int lane) {
  const __bf16* p = base + (size_t)(lane & 15) * ldk + ((lane >> 4) << 3);
  v16bf a;
#pragma unroll
  for (int j = 0; j < 8; ++j) { a[j] = p[j]; a[8 + j] = p[16 + j]; }
  return a;
}

__device__ __forceinline__ v16bf load_a_bf16_f32src(const float* base, size_t ldk, int lane) {
  const float* p = base + (size_t)(lane & 15) * ldk + ((lane >> 4) << 3);
  v16bf a;
#pragma unroll
  for (int j = 0; j < 8; ++j) { a[j] = (__bf16)p[j]; a[8 + j] = (__bf16)p[16 + j]; }
  return a;
}

// B fragment (32x16 bf16, KxN). Col = lane%16; per-lane K runs match A:
// K = (j>>3)*16 + (lane>>4)*8 + (j&7). Element (k, col) at base[col*ldk + k].
__device__ __forceinline__ v16bf load_b_bf16(const __bf16* base, size_t ldk, int lane) {
  const __bf16* p = base + (size_t)(lane & 15) * ldk + ((lane >> 4) << 3);
  v16bf b;
#pragma unroll
  for (int j = 0; j < 8; ++j) { b[j] = p[j]; b[8 + j] = p[16 + j]; }
  return b;
}

// ---------------------------------------------------------------------------
// Tensor Data Mover: 2D tile (bf16 elements) global -> LDS.
// D# bit layout per CDNA5 ISA §8.3/§8.4: group0 {count=1, lds_addr,
// global_addr, type=2}; group1 {data_size=2B, tensor_dim0/1, tile_dim0/1,
// tensor_dim0_stride}. Groups 2/3 + extra group zero (2D tensor).
// amdgpu-toolchain (clang-23) 6-arg builtin form.
// ---------------------------------------------------------------------------
__device__ __forceinline__ void tdm_load_2d_bf16(unsigned lds_addr, const void* gptr,
                                                 unsigned tensor_d0, unsigned tensor_d1,
                                                 unsigned long long d0_stride,
                                                 unsigned tile_d0, unsigned tile_d1) {
  unsigned long long ga = (unsigned long long)gptr;
  u32x4 g0;
  g0[0] = 1u;                                                // count=1 (user mode)
  g0[1] = lds_addr;                                          // LDS byte address
  g0[2] = (unsigned)(ga & 0xffffffffu);                      // global_addr[31:0]
  g0[3] = (unsigned)((ga >> 32) & 0x1ffffffu) | (2u << 30);  // ga[56:32] | type=2
  i32x8 g1;
  g1[0] = (int)(1u << 16);                                   // data_size=1 -> 2 bytes
  g1[1] = (int)((tensor_d0 & 0xffffu) << 16);                // tensor_dim0[15:0]
  g1[2] = (int)((tensor_d0 >> 16) | ((tensor_d1 & 0xffffu) << 16));
  g1[3] = (int)((tensor_d1 >> 16) | (tile_d0 << 16));        // tile_dim0
  g1[4] = (int)(tile_d1 & 0xffffu);                          // tile_dim1 (tile_dim2=0)
  g1[5] = (int)(d0_stride & 0xffffffffu);                    // tensor_dim0_stride lo
  g1[6] = (int)((d0_stride >> 32) & 0xffffu);                // stride hi (dim1_stride=0)
  g1[7] = 0;
  i32x4 gz = {0, 0, 0, 0};
  i32x8 gz8 = {0, 0, 0, 0, 0, 0, 0, 0};
  __builtin_amdgcn_tensor_load_to_lds(g0, g1, gz, gz, gz8, 0);
}

__device__ __forceinline__ unsigned lds_offset_of(const void* p) {
  // flat LDS address: addr[31:0] is the LDS byte offset (ISA §10.2 aperture map)
  return (unsigned)(uintptr_t)p;
}

// ---------------------------------------------------------------------------
// Kernel 1: fused g/theta/phi projection: out[o,n] = W[o,:] . x[b,:,n] + bias
// theta/phi stored [B][N][I] (bf16), g stored [B][I][N] (bf16).
// grid (N/128, 384/16, B), block 256 (8 waves, one 16-col subtile per wave)
// ---------------------------------------------------------------------------
__global__ __launch_bounds__(256)
void proj_kernel(const float* __restrict__ x,
                 const float* __restrict__ g_w,  const float* __restrict__ g_b,
                 const float* __restrict__ th_w, const float* __restrict__ th_b,
                 const float* __restrict__ ph_w, const float* __restrict__ ph_b,
                 __bf16* __restrict__ g_ws, __bf16* __restrict__ th_ws,
                 __bf16* __restrict__ ph_ws) {
  __shared__ __bf16 xs[128][34];  // [n][k] padded
  const int b     = blockIdx.z;
  const int o0    = blockIdx.y * 16;     // 0..383 across {g, theta, phi}
  const int nbase = blockIdx.x * 128;
  const int wv    = threadIdx.x >> 5;
  const int lane  = threadIdx.x & 31;

  const float* wmat; const float* bias; int proj;
  if (o0 < 128)      { wmat = g_w;  bias = g_b;  proj = 0; }
  else if (o0 < 256) { wmat = th_w; bias = th_b; proj = 1; }
  else               { wmat = ph_w; bias = ph_b; proj = 2; }
  const int oo0 = o0 & 127;

  v8f acc = zero_v8f();
  for (int kc = 0; kc < C_; kc += 32) {
    for (int t = threadIdx.x; t < 32 * 128; t += 256) {
      int kk = t >> 7, nn = t & 127;
      xs[nn][kk] = (__bf16)x[((size_t)b * C_ + kc + kk) * N_ + nbase + nn];
    }
    __syncthreads();
    v16bf a  = load_a_bf16_f32src(wmat + (size_t)oo0 * C_ + kc, C_, lane);
    v16bf bb = load_b_bf16(&xs[wv * 16][0], 34, lane);
    acc = wmma_bf16(a, bb, acc);
    __syncthreads();
  }
  const int n = nbase + wv * 16 + (lane & 15);
#pragma unroll
  for (int r = 0; r < 8; ++r) {
    int oo = oo0 + r + ((lane >> 4) << 3);
    __bf16 bv = (__bf16)(acc[r] + bias[oo]);
    if (proj == 0)      g_ws [((size_t)b * I_ + oo) * N_ + n] = bv;  // [B][I][N]
    else if (proj == 1) th_ws[((size_t)b * N_ + n) * I_ + oo] = bv;  // [B][N][I]
    else                ph_ws[((size_t)b * N_ + n) * I_ + oo] = bv;  // [B][N][I]
  }
}

// ---------------------------------------------------------------------------
// Kernel 2: flash attention with TDM double-buffered phi/g tiles in LDS.
// Per wave: 16 query rows; per block all 8 waves share each 64-col phi/g tile
// (eliminates 8x redundant L2 reads). grid (N/128, B), block 256.
// ---------------------------------------------------------------------------
struct AttnSmem {
  __bf16 ph[2][64][128];   // [buf][m_local][i]   2 x 16 KB
  __bf16 gt[2][128][64];   // [buf][i][m_local]   2 x 16 KB
  __bf16 p[8][16][64];     // per-wave P staging  16 KB
};

__global__ __launch_bounds__(256)
void attn_kernel(const __bf16* __restrict__ th, const __bf16* __restrict__ ph,
                 const __bf16* __restrict__ g, __bf16* __restrict__ y) {
  __shared__ AttnSmem sm;
  const int b    = blockIdx.y;
  const int wv   = threadIdx.x >> 5;
  const int lane = threadIdx.x & 31;
  const int wv_u = __builtin_amdgcn_readfirstlane((int)threadIdx.x) >> 5;  // scalar
  const int n0   = blockIdx.x * 128 + wv * 16;

  // theta A-fragments (K = I = 128) are loop-invariant: hoist
  v16bf aTh[4];
#pragma unroll
  for (int kc = 0; kc < 4; ++kc)
    aTh[kc] = load_a_bf16(th + ((size_t)b * N_ + n0) * I_ + kc * 32, I_, lane);

  v8f O[8]; float m[8], l[8];
#pragma unroll
  for (int i = 0; i < 8; ++i) { O[i] = zero_v8f(); m[i] = -1e30f; l[i] = 0.0f; }

  if (wv_u == 0) {  // prime buffer 0 (TDM ignores EXEC; scalar branch via rfl)
    tdm_load_2d_bf16(lds_offset_of(&sm.ph[0][0][0]),
                     ph + (size_t)b * N_ * I_, I_, N_, I_, I_, 64);
    tdm_load_2d_bf16(lds_offset_of(&sm.gt[0][0][0]),
                     g + (size_t)b * I_ * N_, N_, I_, N_, 64, I_);
  }

  for (int mt = 0; mt < N_; mt += 64) {
    const int cur = (mt >> 6) & 1;
    if (wv_u == 0) {
      if (mt + 64 < N_) {
        // prefetch next tile pair into the other buffer, then wait for the
        // current pair (TENSORcnt is in-order: <=2 left => current pair done)
        tdm_load_2d_bf16(lds_offset_of(&sm.ph[cur ^ 1][0][0]),
                         ph + ((size_t)b * N_ + mt + 64) * I_, I_, N_, I_, I_, 64);
        tdm_load_2d_bf16(lds_offset_of(&sm.gt[cur ^ 1][0][0]),
                         g + (size_t)b * I_ * N_ + mt + 64, N_, I_, N_, 64, I_);
        __builtin_amdgcn_s_wait_tensorcnt(2);
      } else {
        __builtin_amdgcn_s_wait_tensorcnt(0);
      }
    }
    __syncthreads();  // current tile visible to all waves

    // S(16x64) = theta_rows . phi_cols, K=128, fragments from LDS
    v8f S[4];
#pragma unroll
    for (int t = 0; t < 4; ++t) {
      S[t] = zero_v8f();
#pragma unroll
      for (int kc = 0; kc < 4; ++kc) {
        v16bf bp = load_b_bf16(&sm.ph[cur][t * 16][kc * 32], 128, lane);
        S[t] = wmma_bf16(aTh[kc], bp, S[t]);
      }
    }
    // online softmax: row = r + 8*(lane/16); columns across 16-lane halves
    float rmax[8], rsum[8], alpha[8];
#pragma unroll
    for (int r = 0; r < 8; ++r) {
      float v = S[0][r];
#pragma unroll
      for (int t = 1; t < 4; ++t) v = fmaxf(v, S[t][r]);
      rmax[r] = v;
    }
#pragma unroll
    for (int mask = 1; mask <= 8; mask <<= 1)
#pragma unroll
      for (int r = 0; r < 8; ++r)
        rmax[r] = fmaxf(rmax[r], __shfl_xor(rmax[r], mask, 32));
#pragma unroll
    for (int r = 0; r < 8; ++r) {
      float mn = fmaxf(m[r], rmax[r]);
      alpha[r] = __expf(m[r] - mn);
      m[r] = mn; rsum[r] = 0.0f;
    }
#pragma unroll
    for (int t = 0; t < 4; ++t)
#pragma unroll
      for (int r = 0; r < 8; ++r) {
        float p = __expf(S[t][r] - m[r]);
        S[t][r] = p; rsum[r] += p;
      }
#pragma unroll
    for (int mask = 1; mask <= 8; mask <<= 1)
#pragma unroll
      for (int r = 0; r < 8; ++r) rsum[r] += __shfl_xor(rsum[r], mask, 32);
#pragma unroll
    for (int r = 0; r < 8; ++r) l[r] = l[r] * alpha[r] + rsum[r];
#pragma unroll
    for (int ci = 0; ci < 8; ++ci)
#pragma unroll
      for (int r = 0; r < 8; ++r) O[ci][r] *= alpha[r];
    // stage P (D-layout -> A-layout) through per-wave LDS region
#pragma unroll
    for (int t = 0; t < 4; ++t)
#pragma unroll
      for (int r = 0; r < 8; ++r)
        sm.p[wv][r + ((lane >> 4) << 3)][t * 16 + (lane & 15)] = (__bf16)S[t][r];
    // O(16x128) += P(16x64) x G(64x128), G fragments from LDS tile
#pragma unroll
    for (int kc2 = 0; kc2 < 2; ++kc2) {
      v16bf aP = load_a_bf16(&sm.p[wv][0][kc2 * 32], 64, lane);
#pragma unroll
      for (int ci = 0; ci < 8; ++ci) {
        v16bf bg = load_b_bf16(&sm.gt[cur][ci * 16][kc2 * 32], 64, lane);
        O[ci] = wmma_bf16(aP, bg, O[ci]);
      }
    }
    __syncthreads();  // all waves done reading buf `cur` before TDM refills it
  }
  float inv[8];
#pragma unroll
  for (int r = 0; r < 8; ++r) inv[r] = 1.0f / l[r];
#pragma unroll
  for (int ci = 0; ci < 8; ++ci)
#pragma unroll
    for (int r = 0; r < 8; ++r) {
      int n = n0 + r + ((lane >> 4) << 3);
      y[((size_t)b * N_ + n) * I_ + ci * 16 + (lane & 15)] =
          (__bf16)(O[ci][r] * inv[r]);
    }
}

// ---------------------------------------------------------------------------
// Kernel 3: wy = w_w x y^T + w_b, plus per-channel sum/sumsq for BN (atomics).
// grid (N/128, C/16, B), block 256.
// ---------------------------------------------------------------------------
__global__ __launch_bounds__(256)
void wgemm_kernel(const float* __restrict__ w_w, const float* __restrict__ w_b,
                  const __bf16* __restrict__ y, float* __restrict__ wy,
                  float* __restrict__ ssum, float* __restrict__ ssq) {
  const int b    = blockIdx.z;
  const int c0   = blockIdx.y * 16;
  const int wv   = threadIdx.x >> 5;
  const int lane = threadIdx.x & 31;
  const int n0   = blockIdx.x * 128 + wv * 16;

  v8f acc = zero_v8f();
#pragma unroll
  for (int kc = 0; kc < 4; ++kc) {
    v16bf a  = load_a_bf16_f32src(w_w + (size_t)c0 * I_ + kc * 32, I_, lane);
    v16bf bb = load_b_bf16(y + ((size_t)b * N_ + n0) * I_ + kc * 32, I_, lane);
    acc = wmma_bf16(a, bb, acc);
  }
  float sv[8], sq[8];
#pragma unroll
  for (int r = 0; r < 8; ++r) {
    int c = c0 + r + ((lane >> 4) << 3);
    float v = acc[r] + w_b[c];
    wy[((size_t)b * C_ + c) * N_ + n0 + (lane & 15)] = v;
    sv[r] = v; sq[r] = v * v;
  }
#pragma unroll
  for (int mask = 1; mask <= 8; mask <<= 1)
#pragma unroll
    for (int r = 0; r < 8; ++r) {
      sv[r] += __shfl_xor(sv[r], mask, 32);
      sq[r] += __shfl_xor(sq[r], mask, 32);
    }
  if ((lane & 15) == 0) {
#pragma unroll
    for (int r = 0; r < 8; ++r) {
      int c = c0 + r + ((lane >> 4) << 3);
      atomicAdd(&ssum[c], sv[r]);
      atomicAdd(&ssq[c], sq[r]);
    }
  }
}

// ---------------------------------------------------------------------------
// Kernel 4: BN finalize (batch stats) + gamma/beta + residual
// ---------------------------------------------------------------------------
__global__ __launch_bounds__(256)
void bn_kernel(const float* __restrict__ wy, const float* __restrict__ ssum,
               const float* __restrict__ ssq, const float* __restrict__ gamma,
               const float* __restrict__ beta, const float* __restrict__ x,
               float* __restrict__ out) {
  size_t idx = (size_t)blockIdx.x * 256 + threadIdx.x;
  int c = (int)((idx >> 12) & (C_ - 1));  // N = 4096 = 2^12
  const float cnt = (float)B_ * (float)N_;
  float mean = ssum[c] / cnt;
  float var  = ssq[c] / cnt - mean * mean;
  out[idx] = (wy[idx] - mean) * rsqrtf(var + BN_EPS) * gamma[c] + beta[c] + x[idx];
}

__global__ void zero_kernel(float* __restrict__ p, int n) {
  int i = blockIdx.x * 256 + threadIdx.x;
  if (i < n) p[i] = 0.0f;
}

extern "C" void kernel_launch(void* const* d_in, const int* in_sizes, int n_in,
                              void* d_out, int out_size, void* d_ws, size_t ws_size,
                              hipStream_t stream) {
  const float* x     = (const float*)d_in[0];
  const float* g_w   = (const float*)d_in[1];
  const float* g_b   = (const float*)d_in[2];
  const float* th_w  = (const float*)d_in[3];
  const float* th_b  = (const float*)d_in[4];
  const float* ph_w  = (const float*)d_in[5];
  const float* ph_b  = (const float*)d_in[6];
  const float* w_w   = (const float*)d_in[7];
  const float* w_b   = (const float*)d_in[8];
  const float* gamma = (const float*)d_in[9];
  const float* beta  = (const float*)d_in[10];
  float* out = (float*)d_out;

  char* ws = (char*)d_ws;
  size_t off = 0;
  const size_t szBNI = (size_t)B_ * N_ * I_ * sizeof(__bf16);           // 8 MB each
  __bf16* g_ws  = (__bf16*)(ws + off); off += szBNI;                    // [B][I][N]
  __bf16* th_ws = (__bf16*)(ws + off); off += szBNI;                    // [B][N][I]
  __bf16* ph_ws = (__bf16*)(ws + off); off += szBNI;                    // [B][N][I]
  __bf16* y_ws  = (__bf16*)(ws + off); off += szBNI;                    // [B][N][I]
  float*  wy_ws = (float*)(ws + off);  off += (size_t)B_ * C_ * N_ * 4; // 33.5 MB
  float*  ssum  = (float*)(ws + off);  off += C_ * sizeof(float);
  float*  ssq   = (float*)(ws + off);  off += C_ * sizeof(float);

  zero_kernel<<<2, 256, 0, stream>>>(ssum, 2 * C_);  // ssum+ssq contiguous
  proj_kernel<<<dim3(N_ / 128, 24, B_), 256, 0, stream>>>(
      x, g_w, g_b, th_w, th_b, ph_w, ph_b, g_ws, th_ws, ph_ws);
  attn_kernel<<<dim3(N_ / 128, B_), 256, 0, stream>>>(th_ws, ph_ws, g_ws, y_ws);
  wgemm_kernel<<<dim3(N_ / 128, C_ / 16, B_), 256, 0, stream>>>(
      w_w, w_b, y_ws, wy_ws, ssum, ssq);
  bn_kernel<<<(B_ * C_ * N_) / 256, 256, 0, stream>>>(
      wy_ws, ssum, ssq, gamma, beta, x, out);
}
